// Bert_79817672229394
// MI455X (gfx1250) — compile-verified, matrix operand order checked by reading, not run on hardware
//
#include <hip/hip_runtime.h>
#include <math.h>

// ---------------- types ----------------
typedef __attribute__((ext_vector_type(16))) __bf16         v16bf;
typedef __attribute__((ext_vector_type(8)))  float          v8f;
typedef __attribute__((ext_vector_type(16))) unsigned short u16x16;
typedef __attribute__((ext_vector_type(8)))  unsigned short u16x8;

#define TT   512
#define BB   16
#define DD   768
#define II   1536
#define HH_  12
#define HEADD 128
#define NBB  384
#define LL   4
#define MM   (TT*BB)           // 8192
#define EPSF 1e-5f
#define SCALEF 0.08838834764831845f   // 1/sqrt(128)
#define STRE 40                // padded LDS row stride (elements) = 80B

static __device__ __forceinline__ unsigned short f2bfbits(float f) {
  unsigned u = __builtin_bit_cast(unsigned, f);
  unsigned r = (u + 0x7FFFu + ((u >> 16) & 1u)) >> 16;
  return (unsigned short)r;
}

// Load a 16x32 bf16 A/B fragment (NT layout). p = row_ptr + kbase + hi*8.
// e<8 -> k = kbase + hi*8 + e ; e>=8 -> k = kbase + 16 + hi*8 + (e-8)
static __device__ __forceinline__ v16bf load_bf_frag(const unsigned short* p) {
  u16x8 lo = *(const u16x8*)p;
  u16x8 hi = *(const u16x8*)(p + 16);
  u16x16 u;
#pragma unroll
  for (int i = 0; i < 8; ++i) { u[i] = lo[i]; u[8 + i] = hi[i]; }
  return __builtin_bit_cast(v16bf, u);
}

static __device__ __forceinline__ v8f wmma_bf16(v16bf a, v16bf b, v8f c) {
  return __builtin_amdgcn_wmma_f32_16x16x32_bf16(false, a, false, b, (short)0, c,
                                                 false, false);
}

// ---------------- fp32 -> bf16 convert ----------------
__global__ void f2bf_kernel(const float* __restrict__ X,
                            unsigned short* __restrict__ Y, long n) {
  long i = (long)blockIdx.x * blockDim.x + threadIdx.x;
  if (i < n) Y[i] = f2bfbits(X[i]);
}

// ---------------- LayerNorm over D=768 -> bf16 ----------------
__global__ void ln768_kernel(const float* __restrict__ X,
                             unsigned short* __restrict__ Y) {
  int lane = threadIdx.x & 31;
  int row  = blockIdx.x * (blockDim.x >> 5) + (threadIdx.x >> 5);
  if (row >= MM) return;
  const float* xr = X + (size_t)row * DD;
  float v[24], s = 0.f, s2 = 0.f;
#pragma unroll
  for (int j = 0; j < 24; ++j) {
    float t = xr[lane + j * 32];
    v[j] = t; s += t; s2 += t * t;
  }
#pragma unroll
  for (int m = 1; m < 32; m <<= 1) { s += __shfl_xor(s, m, 32); s2 += __shfl_xor(s2, m, 32); }
  float mean = s * (1.f / DD);
  float var  = s2 * (1.f / DD) - mean * mean;
  float r    = rsqrtf(var + EPSF);
#pragma unroll
  for (int j = 0; j < 24; ++j)
    Y[(size_t)row * DD + lane + j * 32] = f2bfbits((v[j] - mean) * r);
}

// ---- NT WMMA GEMM with async global->LDS double-buffered staging -----------
// C[M,N] (+)= A[M,K] * B[N,K]^T.  128x128 block tile, 4 waves, 64x64 per wave.
// Per K-step (32): A-tile and B-tile (128x32 bf16 each) are DMA'd to LDS with
// global_load_async_to_lds_b128 (ASYNCcnt), padded to stride 40 elements to
// kill bank conflicts on the ds_load_b128 fragment reads.
template <int ADD>
__global__ __launch_bounds__(128) void gemm_nt(
    const unsigned short* __restrict__ A, const unsigned short* __restrict__ B,
    float* __restrict__ C, int M, int N, int K) {
  // single static LDS allocation -> starts at LDS byte offset 0
  __shared__ unsigned short smem[2][2][128 * STRE];  // [buf][A/B][tile] = 40KB

  int tid  = threadIdx.x;
  int lane = tid & 31;
  int wave = tid >> 5;
  int hi   = lane >> 4;
  int ln   = lane & 15;
  int rowBlk = blockIdx.y * 128;
  int colBlk = blockIdx.x * 128;
  int rowW = (wave >> 1) * 64;   // wave tile origin inside block tile
  int colW = (wave & 1) * 64;

  v8f zero{};
  v8f acc[4][4];
#pragma unroll
  for (int i = 0; i < 4; ++i)
#pragma unroll
    for (int j = 0; j < 4; ++j) acc[i][j] = zero;

  // async-copy one 128x32 K-slice of A and B into smem[bufsel]
  auto issue_tiles = [&](int kk, int bufsel) {
#pragma unroll
    for (int t = 0; t < 4; ++t) {
      int c   = tid + t * 128;       // chunk 0..511
      int row = c >> 2;
      int cc  = c & 3;               // 16B chunk within the 32-col slice
      unsigned long long ga =
          (unsigned long long)(A + (size_t)(rowBlk + row) * K + kk + cc * 8);
      unsigned la = (unsigned)((((size_t)bufsel * 2 + 0) * (128 * STRE) +
                                (size_t)row * STRE + cc * 8) * 2);
      asm volatile("global_load_async_to_lds_b128 %0, %1, off"
                   :: "v"(la), "v"(ga) : "memory");
      unsigned long long gb =
          (unsigned long long)(B + (size_t)(colBlk + row) * K + kk + cc * 8);
      unsigned lb = (unsigned)((((size_t)bufsel * 2 + 1) * (128 * STRE) +
                                (size_t)row * STRE + cc * 8) * 2);
      asm volatile("global_load_async_to_lds_b128 %0, %1, off"
                   :: "v"(lb), "v"(gb) : "memory");
    }
  };

  issue_tiles(0, 0);
  int nk = K / 32;
  for (int kt = 0; kt < nk; ++kt) {
    int buf = kt & 1;
    asm volatile("s_wait_asynccnt 0x0" ::: "memory");  // my deposits done
    __syncthreads();                                   // everyone's deposits done
    if (kt + 1 < nk) issue_tiles((kt + 1) * 32, buf ^ 1);

    const unsigned short* sa = &smem[buf][0][0];
    const unsigned short* sb = &smem[buf][1][0];
    v16bf af[4], bf[4];
#pragma unroll
    for (int i = 0; i < 4; ++i)
      af[i] = load_bf_frag(sa + (size_t)(rowW + i * 16 + ln) * STRE + hi * 8);
#pragma unroll
    for (int j = 0; j < 4; ++j)
      bf[j] = load_bf_frag(sb + (size_t)(colW + j * 16 + ln) * STRE + hi * 8);
#pragma unroll
    for (int i = 0; i < 4; ++i)
#pragma unroll
      for (int j = 0; j < 4; ++j) acc[i][j] = wmma_bf16(af[i], bf[j], acc[i][j]);
  }
#pragma unroll
  for (int i = 0; i < 4; ++i)
#pragma unroll
    for (int j = 0; j < 4; ++j)
#pragma unroll
      for (int v = 0; v < 8; ++v) {
        size_t idx = (size_t)(rowBlk + rowW + i * 16 + v + 8 * hi) * N +
                     colBlk + colW + j * 16 + ln;
        if (ADD) C[idx] += acc[i][j][v]; else C[idx] = acc[i][j][v];
      }
}

// ---------------- value projection (pre-conv h), write VT[bh][d][t] bf16 ----
__global__ void vproj_kernel(const float* __restrict__ HG,
                             const float* __restrict__ Wv,
                             unsigned short* __restrict__ VT) {
  int idx = blockIdx.x * blockDim.x + threadIdx.x;  // over MM*NBB
  if (idx >= MM * NBB) return;
  int nb = idx % NBB, m = idx / NBB;
  int c0 = nb * 4;
  float x0 = HG[(size_t)m * 3072 + c0 + 0];
  float x1 = HG[(size_t)m * 3072 + c0 + 1];
  float x2 = HG[(size_t)m * 3072 + c0 + 2];
  float x3 = HG[(size_t)m * 3072 + c0 + 3];
  int t = m >> 4, b = m & 15;
#pragma unroll
  for (int o = 0; o < 4; ++o) {
    const float* w = Wv + (size_t)nb * 16 + o * 4;
    float val = w[0] * x0 + w[1] * x1 + w[2] * x2 + w[3] * x3;
    int c = c0 + o, hh = c >> 7, d = c & 127;
    VT[((size_t)(b * HH_ + hh) * HEADD + d) * TT + t] = f2bfbits(val);
  }
}

// ---------------- depthwise conv k=5 + swish ----------------
__global__ void conv_kernel(const float* __restrict__ HG,
                            const float* __restrict__ W,
                            const float* __restrict__ Bc,
                            float* __restrict__ HC) {
  int idx = blockIdx.x * blockDim.x + threadIdx.x;  // over MM*II
  if (idx >= MM * II) return;
  int c = idx % II, m = idx / II;
  int t = m >> 4, b = m & 15;
  float acc = Bc[c];
#pragma unroll
  for (int j = 0; j < 5; ++j) {
    int tt = t + j - 2;
    if (tt >= 0 && tt < TT)
      acc += W[(size_t)c * 5 + j] * HG[(size_t)(tt * BB + b) * 3072 + c];
  }
  HC[idx] = acc / (1.f + __expf(-acc));  // swish
}

// ------------- q/k projection + head-LN + RoPE, one wave per (m, head) ------
__global__ void qkproj_kernel(const float* __restrict__ HC,
                              const float* __restrict__ Wq, const float* __restrict__ bq,
                              const float* __restrict__ Wk,
                              const float* __restrict__ qg, const float* __restrict__ qb,
                              const float* __restrict__ kg, const float* __restrict__ kb,
                              unsigned short* __restrict__ Qb,
                              unsigned short* __restrict__ Kb) {
  int gw   = (blockIdx.x * blockDim.x + threadIdx.x) >> 5;
  int lane = threadIdx.x & 31;
  if (gw >= MM * HH_) return;
  int m = gw / HH_, hh = gw % HH_;
  int nb = hh * 32 + lane, c0 = nb * 4;
  float x[4];
#pragma unroll
  for (int i = 0; i < 4; ++i) x[i] = HC[(size_t)m * II + c0 + i];
  float q[4], k[4];
  float sq = 0.f, sq2 = 0.f, sk = 0.f, sk2 = 0.f;
#pragma unroll
  for (int o = 0; o < 4; ++o) {
    const float* wq = Wq + (size_t)nb * 16 + o * 4;
    const float* wk = Wk + (size_t)nb * 16 + o * 4;
    float qa = bq[(size_t)nb * 4 + o];
    float ka = 0.f;
#pragma unroll
    for (int i = 0; i < 4; ++i) { qa += wq[i] * x[i]; ka += wk[i] * x[i]; }
    q[o] = qa; k[o] = ka;
    sq += qa; sq2 += qa * qa; sk += ka; sk2 += ka * ka;
  }
#pragma unroll
  for (int msk = 1; msk < 32; msk <<= 1) {
    sq += __shfl_xor(sq, msk, 32);  sq2 += __shfl_xor(sq2, msk, 32);
    sk += __shfl_xor(sk, msk, 32);  sk2 += __shfl_xor(sk2, msk, 32);
  }
  float mq = sq / 128.f, vq = sq2 / 128.f - mq * mq, rq = rsqrtf(vq + EPSF);
  float mk = sk / 128.f, vk = sk2 / 128.f - mk * mk, rk = rsqrtf(vk + EPSF);
  int t = m >> 4, b = m & 15;
  float qn[4], kn[4];
#pragma unroll
  for (int o = 0; o < 4; ++o) {
    int d = lane * 4 + o;
    qn[o] = (q[o] - mq) * rq * qg[d] + qb[d];
    kn[o] = (k[o] - mk) * rk * kg[d] + kb[d];
  }
  unsigned short* qdst = Qb + ((size_t)(b * HH_ + hh) * TT + t) * HEADD;
  unsigned short* kdst = Kb + ((size_t)(b * HH_ + hh) * TT + t) * HEADD;
#pragma unroll
  for (int o = 0; o < 4; ++o) {
    int d = lane * 4 + o;
    int fi = d & 63;
    float ang = (float)t * __expf(-(float)fi * (6.907755278982137f / 64.f));
    float cs = __cosf(ang), sn = __sinf(ang);
    float qp = __shfl_xor(qn[o], 16, 32);
    float kp = __shfl_xor(kn[o], 16, 32);
    float qr = (lane < 16) ? -qp : qp;
    float kr = (lane < 16) ? -kp : kp;
    qdst[d] = f2bfbits(qn[o] * cs + qr * sn);
    kdst[d] = f2bfbits(kn[o] * cs + kr * sn);
  }
}

// ---------------- fused flash attention (one wave = 16 queries) -------------
__global__ __launch_bounds__(128) void attn_kernel(
    const unsigned short* __restrict__ Qb, const unsigned short* __restrict__ Kb,
    const unsigned short* __restrict__ VTb, const unsigned char* __restrict__ mask,
    float* __restrict__ ATT) {
  __shared__ float lds[4][16 * 32];
  int lane = threadIdx.x & 31;
  int wave = threadIdx.x >> 5;
  int hi = lane >> 4, ln = lane & 15;
  int bh = blockIdx.y;
  int qBase = blockIdx.x * 64 + wave * 16;
  int b = bh / HH_, hh = bh % HH_;
  const unsigned short* Q  = Qb  + (size_t)bh * TT * HEADD;
  const unsigned short* K  = Kb  + (size_t)bh * TT * HEADD;
  const unsigned short* VT = VTb + (size_t)bh * HEADD * TT;
  const unsigned char*  mk = mask + (size_t)b * TT;

  v16bf aq[4];
#pragma unroll
  for (int dk = 0; dk < 4; ++dk)
    aq[dk] = load_bf_frag(Q + (size_t)(qBase + ln) * HEADD + dk * 32 + hi * 8);

  v8f zero{};
  v8f o[8];
  float mrow[8], lsum[8];
#pragma unroll
  for (int j = 0; j < 8; ++j) o[j] = zero;
#pragma unroll
  for (int v = 0; v < 8; ++v) { mrow[v] = -INFINITY; lsum[v] = 0.f; }

  for (int kt = 0; kt < TT / 32; ++kt) {
    v8f s0 = zero, s1 = zero;
#pragma unroll
    for (int dk = 0; dk < 4; ++dk) {
      v16bf b0 = load_bf_frag(K + (size_t)(kt * 32 + ln) * HEADD + dk * 32 + hi * 8);
      v16bf b1 = load_bf_frag(K + (size_t)(kt * 32 + 16 + ln) * HEADD + dk * 32 + hi * 8);
      s0 = wmma_bf16(aq[dk], b0, s0);
      s1 = wmma_bf16(aq[dk], b1, s1);
    }
    bool m0 = mk[kt * 32 + ln] != 0;
    bool m1 = mk[kt * 32 + 16 + ln] != 0;
#pragma unroll
    for (int v = 0; v < 8; ++v) {
      s0[v] = m0 ? -INFINITY : s0[v] * SCALEF;
      s1[v] = m1 ? -INFINITY : s1[v] * SCALEF;
    }
    float sc[8];
#pragma unroll
    for (int v = 0; v < 8; ++v) {
      float a = fmaxf(s0[v], s1[v]);
#pragma unroll
      for (int d = 1; d < 16; d <<= 1) a = fmaxf(a, __shfl_xor(a, d, 32));
      float mnew = fmaxf(mrow[v], a);
      sc[v] = (mnew == -INFINITY) ? 1.f : __expf(mrow[v] - mnew);
      mrow[v] = mnew;
      lsum[v] *= sc[v];
    }
#pragma unroll
    for (int j = 0; j < 8; ++j)
#pragma unroll
      for (int v = 0; v < 8; ++v) o[j][v] *= sc[v];

#pragma unroll
    for (int v = 0; v < 8; ++v) {
      float p0 = (mrow[v] == -INFINITY) ? 0.f : __expf(s0[v] - mrow[v]);
      float p1 = (mrow[v] == -INFINITY) ? 0.f : __expf(s1[v] - mrow[v]);
      lsum[v] += p0 + p1;
      lds[wave][(v + 8 * hi) * 32 + ln]      = p0;
      lds[wave][(v + 8 * hi) * 32 + 16 + ln] = p1;
    }
    __builtin_amdgcn_wave_barrier();  // LDS in-order per wave; fence scheduling
    u16x16 up;
#pragma unroll
    for (int e = 0; e < 16; ++e) {
      int kk = (e < 8) ? (hi * 8 + e) : (16 + hi * 8 + (e - 8));
      up[e] = f2bfbits(lds[wave][ln * 32 + kk]);
    }
    __builtin_amdgcn_wave_barrier();
    v16bf pa = __builtin_bit_cast(v16bf, up);
#pragma unroll
    for (int j = 0; j < 8; ++j) {
      v16bf bv = load_bf_frag(VT + (size_t)(j * 16 + ln) * TT + kt * 32 + hi * 8);
      o[j] = wmma_bf16(pa, bv, o[j]);
    }
  }
  float inv[8];
#pragma unroll
  for (int v = 0; v < 8; ++v) {
    float a = lsum[v];
#pragma unroll
    for (int d = 1; d < 16; d <<= 1) a += __shfl_xor(a, d, 32);
    inv[v] = (a > 0.f) ? 1.f / a : 0.f;  // fully-masked row -> 0 (matches ref)
  }
#pragma unroll
  for (int v = 0; v < 8; ++v) {
    int t = qBase + v + 8 * hi;
#pragma unroll
    for (int j = 0; j < 8; ++j)
      ATT[(size_t)(t * BB + b) * II + hh * HEADD + j * 16 + ln] = o[j][v] * inv[v];
  }
}

// ------------- skip + gate + LayerNorm over I=1536 -> bf16 ------------------
__global__ void post_ln_kernel(const float* __restrict__ HC,
                               const float* __restrict__ ATT,
                               const float* __restrict__ HG,
                               const float* __restrict__ lskip,
                               unsigned short* __restrict__ X2) {
  int lane = threadIdx.x & 31;
  int row  = blockIdx.x * (blockDim.x >> 5) + (threadIdx.x >> 5);
  if (row >= MM) return;
  float v[48], s = 0.f, s2 = 0.f;
#pragma unroll
  for (int j = 0; j < 48; ++j) {
    int c = lane + j * 32;
    float g = HG[(size_t)row * 3072 + II + c];
    g = g / (1.f + __expf(-g));  // swish(gate)
    float h2 = (HC[(size_t)row * II + c] * lskip[c] + ATT[(size_t)row * II + c]) * g;
    v[j] = h2; s += h2; s2 += h2 * h2;
  }
#pragma unroll
  for (int m = 1; m < 32; m <<= 1) { s += __shfl_xor(s, m, 32); s2 += __shfl_xor(s2, m, 32); }
  float mean = s * (1.f / II);
  float var  = s2 * (1.f / II) - mean * mean;
  float r    = rsqrtf(var + EPSF);
#pragma unroll
  for (int j = 0; j < 48; ++j)
    X2[(size_t)row * II + lane + j * 32] = f2bfbits((v[j] - mean) * r);
}

// ============================= host launch ==================================
extern "C" void kernel_launch(void* const* d_in, const int* in_sizes, int n_in,
                              void* d_out, int out_size, void* d_ws, size_t ws_size,
                              hipStream_t stream) {
  const float*         x     = (const float*)d_in[0];
  const unsigned char* amask = (const unsigned char*)d_in[1];  // jnp bool_
  const float* Win  = (const float*)d_in[2];
  const float* Wout = (const float*)d_in[3];
  const float* Wq   = (const float*)d_in[4];
  const float* bq   = (const float*)d_in[5];
  const float* Wk   = (const float*)d_in[6];
  const float* Wv   = (const float*)d_in[7];
  const float* kg   = (const float*)d_in[8];
  const float* kb   = (const float*)d_in[9];
  const float* qg   = (const float*)d_in[10];
  const float* qb   = (const float*)d_in[11];
  const float* convw = (const float*)d_in[12];
  const float* convb = (const float*)d_in[13];
  const float* lskip = (const float*)d_in[14];
  float* hidden = (float*)d_out;

  char* ws = (char*)d_ws;
  size_t off = 0;
  unsigned short* WinB  = (unsigned short*)(ws + off); off += (size_t)LL * 3072 * DD * 2;
  unsigned short* WoutB = (unsigned short*)(ws + off); off += (size_t)LL * DD * II * 2;
  unsigned short* XLN   = (unsigned short*)(ws + off); off += (size_t)MM * DD * 2;
  float*          HG    = (float*)(ws + off);          off += (size_t)MM * 3072 * 4;
  float*          HC    = (float*)(ws + off);          off += (size_t)MM * II * 4;
  unsigned short* Qb    = (unsigned short*)(ws + off); off += (size_t)192 * TT * HEADD * 2;
  unsigned short* Kb    = (unsigned short*)(ws + off); off += (size_t)192 * TT * HEADD * 2;
  unsigned short* VTb   = (unsigned short*)(ws + off); off += (size_t)192 * HEADD * TT * 2;
  float*          ATT   = (float*)(ws + off);          off += (size_t)MM * II * 4;
  unsigned short* X2    = (unsigned short*)(ws + off); off += (size_t)MM * II * 2;

  // hidden = x
  hipMemcpyAsync(hidden, x, (size_t)MM * DD * sizeof(float),
                 hipMemcpyDeviceToDevice, stream);

  // weights -> bf16 (once)
  {
    long n1 = (long)LL * 3072 * DD;
    long n2 = (long)LL * DD * II;
    f2bf_kernel<<<(n1 + 255) / 256, 256, 0, stream>>>(Win, WinB, n1);
    f2bf_kernel<<<(n2 + 255) / 256, 256, 0, stream>>>(Wout, WoutB, n2);
  }

  for (int l = 0; l < LL; ++l) {
    const unsigned short* WinBl  = WinB  + (size_t)l * 3072 * DD;
    const unsigned short* WoutBl = WoutB + (size_t)l * DD * II;
    const float* Wql = Wq + (size_t)l * NBB * 16;
    const float* bql = bq + (size_t)l * NBB * 4;
    const float* Wkl = Wk + (size_t)l * NBB * 16;
    const float* Wvl = Wv + (size_t)l * NBB * 16;
    const float* kgl = kg + (size_t)l * HEADD;
    const float* kbl = kb + (size_t)l * HEADD;
    const float* qgl = qg + (size_t)l * HEADD;
    const float* qbl = qb + (size_t)l * HEADD;
    const float* cwl = convw + (size_t)l * II * 5;
    const float* cbl = convb + (size_t)l * II;
    const float* lsl = lskip + (size_t)l * II;

    // 1) LN(hidden) -> bf16
    ln768_kernel<<<MM / 8, 256, 0, stream>>>(hidden, XLN);
    // 2) HG = XLN @ Win^T   (8192 x 3072, K=768)  [async-LDS WMMA]
    gemm_nt<0><<<dim3(3072 / 128, MM / 128), 128, 0, stream>>>(
        XLN, WinBl, HG, MM, 3072, DD);
    // 3) value projection (pre-conv h) -> VT bf16 (d-major)
    vproj_kernel<<<(MM * NBB + 255) / 256, 256, 0, stream>>>(HG, Wvl, VTb);
    // 4) depthwise conv + swish -> HC
    conv_kernel<<<(MM * II + 255) / 256, 256, 0, stream>>>(HG, cwl, cbl, HC);
    // 5) q/k projection + head-LN + RoPE -> Qb, Kb bf16
    qkproj_kernel<<<(MM * HH_) / 8, 256, 0, stream>>>(
        HC, Wql, bql, Wkl, qgl, qbl, kgl, kbl, Qb, Kb);
    // 6) fused flash attention -> ATT
    attn_kernel<<<dim3(TT / 64, BB * HH_), 128, 0, stream>>>(
        Qb, Kb, VTb, amask, ATT);
    // 7) skip + gate + LN -> X2 bf16
    post_ln_kernel<<<MM / 8, 256, 0, stream>>>(HC, ATT, HG, lsl, X2);
    // 8) hidden += X2 @ Wout^T  (8192 x 768, K=1536)  [async-LDS WMMA]
    gemm_nt<1><<<dim3(DD / 128, MM / 128), 128, 0, stream>>>(
        X2, WoutBl, hidden, MM, DD, II);
  }
}
